// ImplicitKernelAttention_82343112999026
// MI455X (gfx1250) — compile-verified, matrix operand order checked by reading, not run on hardware
//
#include <hip/hip_runtime.h>
#include <hip/hip_bf16.h>

typedef __attribute__((ext_vector_type(16))) _Float16 v16h;
typedef __attribute__((ext_vector_type(8)))  float    v8f;

#define B_   2
#define H_   16
#define S_   2048
#define D_   64
#define R_   64
#define BH_  (B_ * H_)
#define T_   (BH_ * S_)
#define PHI_ (2 * R_)   // 128 features after [cos, sin]

#define LOG2E_ 1.4426950408889634f

struct __align__(16) U4 { unsigned int x, y, z, w; };

// Load a v16h WMMA A/B fragment: 8 halves at p, 8 halves at p+16 (two b128 loads)
__device__ __forceinline__ v16h load_frag16(const _Float16* p) {
  union Pack { v16h h; U4 u[2]; };
  Pack t;
  t.u[0] = *(const U4*)(p);
  t.u[1] = *(const U4*)(p + 16);
  return t.h;
}

// --- Phase 1: random-feature projection ---
// phi = [cos(XW+b), sin(XW+b)] (f16), bfac = exp(||x||^2/16)/256.
// The query-side bfac is never read (the row factor cancels in softmax);
// the key-side bfac is the B_k weight of the normalized-score formulation.
__global__ __launch_bounds__(256)
void ika_proj(const float* __restrict__ X, const float* __restrict__ W,
              const float* __restrict__ bias, _Float16* __restrict__ phi,
              float* __restrict__ bfac) {
  __shared__ float sW[D_ * R_];     // 16 KB
  __shared__ float sX[4][D_];
  __shared__ float red[4][2];
  const int tid = threadIdx.x;
  for (int i = tid; i < D_ * R_; i += 256) sW[i] = W[i];
  const int sub = tid >> 6;                       // token within block (4 per block)
  const int r   = tid & 63;                       // feature index
  const int tok = blockIdx.x * 4 + sub;
  sX[sub][r] = X[(size_t)tok * D_ + r];
  __syncthreads();
  float w  = bias[r];
  float xr = sX[sub][r];
  float nq = xr * xr;
#pragma unroll 8
  for (int d = 0; d < D_; ++d) w += sX[sub][d] * sW[d * R_ + r];
  phi[(size_t)tok * PHI_ + r]      = (_Float16)__cosf(w);
  phi[(size_t)tok * PHI_ + R_ + r] = (_Float16)__sinf(w);
  // reduce ||x||^2 over the 64 threads of this token (2 waves)
  nq += __shfl_xor(nq, 1, 32);
  nq += __shfl_xor(nq, 2, 32);
  nq += __shfl_xor(nq, 4, 32);
  nq += __shfl_xor(nq, 8, 32);
  nq += __shfl_xor(nq, 16, 32);
  if ((tid & 31) == 0) red[sub][(tid >> 5) & 1] = nq;
  __syncthreads();
  if (r == 0)
    bfac[tok] = __builtin_amdgcn_exp2f((red[sub][0] + red[sub][1]) *
                                       (LOG2E_ * 0.0625f)) * (1.0f / 256.0f);
}

// --- Phase 2: V transpose to [bh][d][s] in f16 so PV B-fragments are contiguous ---
__global__ __launch_bounds__(256)
void ika_vt(const float* __restrict__ V, _Float16* __restrict__ Vt) {
  const size_t idx = (size_t)blockIdx.x * 256 + threadIdx.x;   // bh*D*S + d*S + s
  const int    s   = (int)(idx & (S_ - 1));
  const size_t t   = idx >> 11;              // S_ == 2^11
  const int    d   = (int)(t & (D_ - 1));
  const int    bh  = (int)(t >> 6);          // D_ == 2^6
  Vt[idx] = (_Float16)V[((size_t)bh * S_ + s) * D_ + d];
}

// --- Phase 3: fused attention. One wave owns a 32-row Q tile (2 row-halves), ---
// so every phi_k / Vt fragment loaded from L2 feeds two WMMA sets (halved B-traffic).
// attn[q][k] = E_qk * B_k / sum_k(E_qk * B_k), E = (dot/64)^2 + 1e-5.
// No log/exp/max: softmax's row factor cancels algebraically; values are f32-safe.
// Causality: 32-row x 32-key alignment means only the final (diagonal) tile needs
// a mask, and that mask is the lane-local predicate (v + 8*hi >= row).
__global__ __launch_bounds__(128)
void ika_flash(const _Float16* __restrict__ phiQ, const _Float16* __restrict__ phiK,
               const _Float16* __restrict__ Vt, const float* __restrict__ bk,
               float* __restrict__ out) {
  __shared__ _Float16 sP[4][16 * 32];        // per-wave P staging (C-layout -> A-layout)

  const int wave = threadIdx.x >> 5;
  const int lane = threadIdx.x & 31;
  const int row  = lane & 15;                // N column / matrix row (depending on fragment)
  const int hi   = lane >> 4;                // K-half selector / M+8 selector
  const int bh   = blockIdx.x >> 4;          // 16 q-blocks of 128 rows per (b,h)
  const int q0   = (blockIdx.x & 15) * 128 + wave * 32;   // wave rows q0..q0+31
  const size_t tb = (size_t)bh * S_;

  // A fragments: two 16-row halves, K = 0..127 in 4 chunks of 32
  v16h a[2][4];
#pragma unroll
  for (int h = 0; h < 2; ++h) {
    const _Float16* pq = phiQ + (tb + q0 + 16 * h + row) * PHI_ + hi * 8;
#pragma unroll
    for (int c = 0; c < 4; ++c) a[h][c] = load_frag16(pq + c * 32);
  }

  // constant all-ones B fragment: row-sums of P via WMMA (P x 1)
  v16h onesb;
#pragma unroll
  for (int i = 0; i < 16; ++i) onesb[i] = (_Float16)1.0f;

  const v8f zero = {0.f, 0.f, 0.f, 0.f, 0.f, 0.f, 0.f, 0.f};
  v8f o[2][4] = {{zero, zero, zero, zero}, {zero, zero, zero, zero}};
  v8f lsum[2] = {zero, zero};

  const int ktiles = (q0 >> 5) + 1;          // last tile (k0 == q0) is the diagonal
  for (int kt = 0; kt < ktiles; ++kt) {
    const int k0 = kt * 32;
    const _Float16* pk0 = phiK + (tb + k0 + row) * PHI_ + hi * 8;
    const _Float16* pk1 = pk0 + 16 * PHI_;

    // prefetch next tile's phi_k rows toward the WGP (global_prefetch_b8)
    if (kt + 1 < ktiles) {
      __builtin_prefetch(pk0 + 32 * PHI_, 0, 3);
      __builtin_prefetch(pk1 + 32 * PHI_, 0, 3);
    }

    // scores: 2 row-halves x 2 key-subtiles; each B fragment feeds both halves
    v8f s[2][2] = {{zero, zero}, {zero, zero}};
#pragma unroll
    for (int c = 0; c < 4; ++c) {
      v16h b0 = load_frag16(pk0 + c * 32);
      v16h b1 = load_frag16(pk1 + c * 32);
      s[0][0] = __builtin_amdgcn_wmma_f32_16x16x32_f16(false, a[0][c], false, b0,
                                                       (short)0, s[0][0], false, false);
      s[1][0] = __builtin_amdgcn_wmma_f32_16x16x32_f16(false, a[1][c], false, b0,
                                                       (short)0, s[1][0], false, false);
      s[0][1] = __builtin_amdgcn_wmma_f32_16x16x32_f16(false, a[0][c], false, b1,
                                                       (short)0, s[0][1], false, false);
      s[1][1] = __builtin_amdgcn_wmma_f32_16x16x32_f16(false, a[1][c], false, b1,
                                                       (short)0, s[1][1], false, false);
    }

    // per-column constants: w = s^2 * (ck/4096) + ck*1e-5
    const float ck0 = bk[tb + k0 + row];
    const float ck1 = bk[tb + k0 + 16 + row];
    const float cm0 = ck0 * (1.0f / 4096.0f), ca0 = ck0 * 1e-5f;
    const float cm1 = ck1 * (1.0f / 4096.0f), ca1 = ck1 * 1e-5f;

    if (kt + 1 < ktiles) {
      // bulk path: fully below the diagonal, no mask
#pragma unroll
      for (int v = 0; v < 8; ++v) {
        s[0][0][v] = s[0][0][v] * s[0][0][v] * cm0 + ca0;
        s[1][0][v] = s[1][0][v] * s[1][0][v] * cm0 + ca0;
        s[0][1][v] = s[0][1][v] * s[0][1][v] * cm1 + ca1;
        s[1][1][v] = s[1][1][v] * s[1][1][v] * cm1 + ca1;
      }
    } else {
      // diagonal tile: [h0][t1] fully masked, [h1][t0] fully open,
      // [h0][t0] and [h1][t1] use the lane-local predicate (v+8*hi >= row)
#pragma unroll
      for (int v = 0; v < 8; ++v) {
        const bool dg = (v + 8 * hi) >= row;
        float w00 = s[0][0][v] * s[0][0][v] * cm0 + ca0;
        s[0][0][v] = dg ? w00 : 0.0f;
        s[0][1][v] = 0.0f;
        s[1][0][v] = s[1][0][v] * s[1][0][v] * cm0 + ca0;
        float w11 = s[1][1][v] * s[1][1][v] * cm1 + ca1;
        s[1][1][v] = dg ? w11 : 0.0f;
      }
    }

    // P (C layout) -> LDS row-major 16x32 -> reload in A layout, per half.
    // Same-wave LDS ops execute in order, so the half-0 load drains before
    // half-1's stores overwrite the buffer.
    v16h pa[2];
    _Float16* lp = &sP[wave][0];
#pragma unroll
    for (int h = 0; h < 2; ++h) {
#pragma unroll
      for (int v = 0; v < 8; ++v) {
        lp[(v + 8 * hi) * 32 + row]      = (_Float16)s[h][0][v];
        lp[(v + 8 * hi) * 32 + row + 16] = (_Float16)s[h][1][v];
      }
      pa[h] = load_frag16(lp + row * 32 + hi * 8);
      lsum[h] = __builtin_amdgcn_wmma_f32_16x16x32_f16(false, pa[h], false, onesb,
                                                       (short)0, lsum[h], false, false);
    }

    // O += P(16x32) x V(32x64): each Vt fragment feeds both row-halves
#pragma unroll
    for (int c = 0; c < 4; ++c) {
      const _Float16* pv = Vt + ((size_t)bh * D_ + row + 16 * c) * S_ + k0 + hi * 8;
      v16h bv = load_frag16(pv);
      o[0][c] = __builtin_amdgcn_wmma_f32_16x16x32_f16(false, pa[0], false, bv,
                                                       (short)0, o[0][c], false, false);
      o[1][c] = __builtin_amdgcn_wmma_f32_16x16x32_f16(false, pa[1], false, bv,
                                                       (short)0, o[1][c], false, false);
    }
  }

  // epilogue: normalize by row sum (v_rcp_f32), write out [B,H,S,D] f32
#pragma unroll
  for (int h = 0; h < 2; ++h) {
#pragma unroll
    for (int v = 0; v < 8; ++v) {
      const float inv = __builtin_amdgcn_rcpf(lsum[h][v]);
      float* po = out + (tb + q0 + 16 * h + v + 8 * hi) * D_;
#pragma unroll
      for (int c = 0; c < 4; ++c) po[row + 16 * c] = o[h][c][v] * inv;
    }
  }
}

extern "C" void kernel_launch(void* const* d_in, const int* in_sizes, int n_in,
                              void* d_out, int out_size, void* d_ws, size_t ws_size,
                              hipStream_t stream) {
  (void)in_sizes; (void)n_in; (void)out_size; (void)ws_size;
  const float* query = (const float*)d_in[0];
  const float* keys  = (const float*)d_in[1];
  const float* vals  = (const float*)d_in[2];
  // d_in[3] = mask: unused, causal structure is computed analytically
  const float* W     = (const float*)d_in[4];
  const float* bias  = (const float*)d_in[5];
  float*       out   = (float*)d_out;

  // workspace layout (40.5 MB total)
  char* ws = (char*)d_ws;
  _Float16* phiQ = (_Float16*)(ws);                                   // 16 MB
  _Float16* phiK = (_Float16*)(ws + (size_t)16 * 1024 * 1024);        // 16 MB
  _Float16* Vt   = (_Float16*)(ws + (size_t)32 * 1024 * 1024);        //  8 MB
  float*    bq   = (float*)   (ws + (size_t)40 * 1024 * 1024);        // 256 KB (unused)
  float*    bkb  = (float*)   (ws + (size_t)40 * 1024 * 1024 + 256 * 1024);

  ika_proj<<<T_ / 4, 256, 0, stream>>>(query, W, bias, phiQ, bq);
  ika_proj<<<T_ / 4, 256, 0, stream>>>(keys,  W, bias, phiK, bkb);
  ika_vt<<<(BH_ * D_ * S_) / 256, 256, 0, stream>>>(vals, Vt);
  ika_flash<<<BH_ * (S_ / 128), 128, 0, stream>>>(phiQ, phiK, Vt, bkb, out);
}
